// FSCTEncoderPyG_13237089206893
// MI455X (gfx1250) — matrix operand order repaired and checked by reading, not compiled, not run on hardware
//
#include <hip/hip_runtime.h>
#include <hip/hip_bf16.h>
#include <stdint.h>

// ---------------------------------------------------------------------------
// PointNet++ (FSCT) encoder for MI455X / gfx1250.
// GEMMs: v_wmma_f32_16x16x32_f16 (wave32 WMMA), double-buffered
// global_load_async_to_lds_b128 staging (ASYNCcnt), fused bias+ReLU+BN
// epilogue. Weights pre-transposed to [Nc][KaPad], KaPad % 64 == 0.
// ---------------------------------------------------------------------------

typedef __attribute__((ext_vector_type(16))) _Float16 v16h;
typedef __attribute__((ext_vector_type(8)))  _Float16 h8;
typedef __attribute__((ext_vector_type(8)))  float    v8f;

#define NPTS  32768
#define M1    3277
#define M2    164
#define KNBR  64
#define BN_INV 0.9999950000374997f   // 1/sqrt(1+1e-5)

// ---------------------------------------------------------------------------
// pos -> out[x], out[pos]
__global__ void k_copy_pos(const float* __restrict__ pos, float* __restrict__ out)
{
    int g = blockIdx.x * blockDim.x + threadIdx.x;
    if (g < NPTS * 3) {
        float v = pos[g];
        out[g] = v;               // x
        out[NPTS * 3 + g] = v;    // pos
    }
}

// ---------------------------------------------------------------------------
// Farthest point sampling: single block, 1024 threads, LDS tree argmax.
__global__ __launch_bounds__(1024) void k_fps(const float* __restrict__ pos,
                                              int n, int m,
                                              int* __restrict__ out_idx,
                                              float* __restrict__ dist)
{
    __shared__ float sv[1024];
    __shared__ int   si[1024];
    __shared__ int   slast;
    const int tid = threadIdx.x;

    for (int i = tid; i < n; i += 1024) dist[i] = 3.4e38f;
    if (tid == 0) { out_idx[0] = 0; slast = 0; }
    __syncthreads();

    for (int s = 1; s < m; ++s) {
        const int last = slast;
        const float px = pos[last * 3 + 0];
        const float py = pos[last * 3 + 1];
        const float pz = pos[last * 3 + 2];
        float bv = -1.0f; int ba = 0;
        for (int i = tid; i < n; i += 1024) {
            float dx = pos[i * 3 + 0] - px;
            float dy = pos[i * 3 + 1] - py;
            float dz = pos[i * 3 + 2] - pz;
            float d  = dx * dx + dy * dy + dz * dz;
            float nd = fminf(dist[i], d);
            dist[i] = nd;
            if (nd > bv || (nd == bv && i < ba)) { bv = nd; ba = i; }
        }
        sv[tid] = bv; si[tid] = ba;
        __syncthreads();
        for (int off = 512; off > 0; off >>= 1) {
            if (tid < off) {
                float v2 = sv[tid + off]; int i2 = si[tid + off];
                if (v2 > sv[tid] || (v2 == sv[tid] && i2 < si[tid])) { sv[tid] = v2; si[tid] = i2; }
            }
            __syncthreads();
        }
        if (tid == 0) { out_idx[s] = si[0]; slast = si[0]; }
        __syncthreads();
    }
}

// ---------------------------------------------------------------------------
// gather centers = src[idx]
__global__ void k_gather3(const float* __restrict__ src, const int* __restrict__ idx,
                          int m, float* __restrict__ out)
{
    int g = blockIdx.x * blockDim.x + threadIdx.x;
    if (g < m * 3) {
        int j = g / 3, a = g - j * 3;
        out[g] = src[idx[j] * 3 + a];
    }
}

// ---------------------------------------------------------------------------
// deterministic radius neighbors: self first, then first (K-1) points with
// d2 <= r2 in index order; pad with -1.
__global__ void k_radius(const float* __restrict__ src, int n,
                         const float* __restrict__ centers,
                         const int* __restrict__ cidx, int m,
                         float r2, int* __restrict__ nbr)
{
    int g = blockIdx.x * blockDim.x + threadIdx.x;
    if (g >= m) return;
    const float cx = centers[g * 3 + 0];
    const float cy = centers[g * 3 + 1];
    const float cz = centers[g * 3 + 2];
    const int self = cidx[g];
    int* row = nbr + (size_t)g * KNBR;
    row[0] = self;
    int cnt = 1;
    for (int i = 0; i < n && cnt < KNBR; ++i) {
        if (i == self) continue;
        float dx = src[i * 3 + 0] - cx;
        float dy = src[i * 3 + 1] - cy;
        float dz = src[i * 3 + 2] - cz;
        if (dx * dx + dy * dy + dz * dz <= r2) row[cnt++] = i;
    }
    for (; cnt < KNBR; ++cnt) row[cnt] = -1;
}

// ---------------------------------------------------------------------------
// f32 weights (Cin x Cout) -> f16 TRANSPOSED (Cout x KaPad), zero-padded k.
__global__ void k_convw_t(const float* __restrict__ W, int Cin, int Cout, int KaPad,
                          _Float16* __restrict__ outT)
{
    int g = blockIdx.x * blockDim.x + threadIdx.x;
    if (g < KaPad * Cout) {
        int n = g / KaPad, k = g - n * KaPad;
        outT[g] = (k < Cin) ? (_Float16)W[(size_t)k * Cout + n] : (_Float16)0.0f;
    }
}

// ---------------------------------------------------------------------------
// edge feature gather: feat[m*K+k, :] = [x_src[nbr], pos_src[nbr]-center, 0pad]
__global__ void k_feat(const float* __restrict__ xsrc, int Cx,
                       const float* __restrict__ psrc,
                       const float* __restrict__ centers,
                       const int* __restrict__ nbr,
                       int M, int KaPad, _Float16* __restrict__ feat)
{
    size_t g = (size_t)blockIdx.x * blockDim.x + threadIdx.x;
    size_t total = (size_t)M * KNBR * KaPad;
    if (g >= total) return;
    int row = (int)(g / KaPad);
    int col = (int)(g - (size_t)row * KaPad);
    int m = row >> 6;
    int j = nbr[row];
    float v = 0.0f;
    if (j >= 0) {
        if (col < Cx)          v = xsrc[(size_t)j * Cx + col];
        else if (col < Cx + 3) { int a = col - Cx; v = psrc[j * 3 + a] - centers[m * 3 + a]; }
    }
    feat[g] = (_Float16)v;
}

// SA3 input: feat[m, :] = [x2[m], c2[m], 0pad]   (164 x 1088)
__global__ void k_feat3(const float* __restrict__ x2, const float* __restrict__ c2,
                        _Float16* __restrict__ feat)
{
    int g = blockIdx.x * blockDim.x + threadIdx.x;
    if (g >= M2 * 1088) return;
    int row = g / 1088, col = g - row * 1088;
    float v = 0.0f;
    if (col < 1024)      v = x2[(size_t)row * 1024 + col];
    else if (col < 1027) v = c2[row * 3 + (col - 1024)];
    feat[g] = (_Float16)v;
}

// ---------------------------------------------------------------------------
// WMMA GEMM + fused  y = gamma*inv * relu(A@Wt^T + b) + beta,  f16 out.
// A: [M][Ka] row-major f16.  Wt: [Nc][Ka] (pre-transposed weights).
// Tile 128x128x64, 512 threads = 16 waves, each wave: 2x2 frags of 16x16,
// 8 WMMA per stage. Double-buffered GLOBAL_LOAD_ASYNC_TO_LDS_B128 staging:
// tile t+1 is in flight while tile t is consumed (s_wait_asynccnt 0x4).
#define BM 128
#define BNT 128
#define BK 64
__global__ __launch_bounds__(512)
void k_gemm_wmma(const _Float16* __restrict__ A, const _Float16* __restrict__ Wt,
                 const float* __restrict__ bias, const float* __restrict__ gamma,
                 const float* __restrict__ beta, _Float16* __restrict__ out,
                 int M, int Ka, int Nc)
{
    __shared__ alignas(64) _Float16 As[2][BM * BK];    // 2 x 16 KB, [r][k]
    __shared__ alignas(64) _Float16 Bst[2][BNT * BK];  // 2 x 16 KB, [n][k]

    const int tid  = threadIdx.x;
    const int wave = tid >> 5;
    const int lane = tid & 31;
    const int wm   = wave >> 2;        // 0..3  (row block of 32)
    const int wn   = wave & 3;         // 0..3  (col block of 32)
    const int lr   = lane & 15;
    const int hi   = lane >> 4;        // 0/1
    const int bm   = blockIdx.x * BM;
    const int bn   = blockIdx.y * BNT;

    // staging assignment: each thread copies two 16B chunks of A and of B.
    // chunk c -> row c>>3 (of 128), quad c&7 (8 x 8 halves per 64-half row)
    const int r = tid >> 3, q = tid & 7;          // r: 0..63, q: 0..7
    const int gra = min(bm + r,      M - 1);      // clamped (dup rows discarded
    const int grb = min(bm + r + 64, M - 1);      //  by epilogue row<M guard)
    const _Float16* gpa0 = A  + (size_t)gra * Ka + q * 8;
    const _Float16* gpa1 = A  + (size_t)grb * Ka + q * 8;
    const _Float16* gpb0 = Wt + (size_t)(bn + r) * Ka + q * 8;       // Nc%128==0
    const _Float16* gpb1 = Wt + (size_t)(bn + r + 64) * Ka + q * 8;
    uint32_t la0[2], la1[2], lb0[2], lb1[2];
#pragma unroll
    for (int b = 0; b < 2; ++b) {
        la0[b] = (uint32_t)(uintptr_t)(&As[b][(r)      * BK + q * 8]);
        la1[b] = (uint32_t)(uintptr_t)(&As[b][(r + 64) * BK + q * 8]);
        lb0[b] = (uint32_t)(uintptr_t)(&Bst[b][(r)      * BK + q * 8]);
        lb1[b] = (uint32_t)(uintptr_t)(&Bst[b][(r + 64) * BK + q * 8]);
    }

    v8f acc[2][2] = {};
    const int nk = Ka >> 6;            // Ka % 64 == 0

    // prologue: stage tile 0 into buffer 0
    {
        asm volatile("global_load_async_to_lds_b128 %0, %1, off" :: "v"(la0[0]), "v"(gpa0) : "memory");
        asm volatile("global_load_async_to_lds_b128 %0, %1, off" :: "v"(la1[0]), "v"(gpa1) : "memory");
        asm volatile("global_load_async_to_lds_b128 %0, %1, off" :: "v"(lb0[0]), "v"(gpb0) : "memory");
        asm volatile("global_load_async_to_lds_b128 %0, %1, off" :: "v"(lb1[0]), "v"(gpb1) : "memory");
    }

    for (int t = 0; t < nk; ++t) {
        const int cur = t & 1;
        if (t + 1 < nk) {
            const int kb = (t + 1) << 6;
            const int nxt = cur ^ 1;
            asm volatile("global_load_async_to_lds_b128 %0, %1, off" :: "v"(la0[nxt]), "v"(gpa0 + kb) : "memory");
            asm volatile("global_load_async_to_lds_b128 %0, %1, off" :: "v"(la1[nxt]), "v"(gpa1 + kb) : "memory");
            asm volatile("global_load_async_to_lds_b128 %0, %1, off" :: "v"(lb0[nxt]), "v"(gpb0 + kb) : "memory");
            asm volatile("global_load_async_to_lds_b128 %0, %1, off" :: "v"(lb1[nxt]), "v"(gpb1 + kb) : "memory");
            if (t + 2 < nk) {   // warm L2/WGP$ two stages ahead
                __builtin_prefetch(gpa0 + kb + BK, 0, 3);
                __builtin_prefetch(gpb0 + kb + BK, 0, 3);
            }
            // just-issued 4 may remain outstanding; tile t's 4 must be done
            asm volatile("s_wait_asynccnt 0x4" ::: "memory");
        } else {
            asm volatile("s_wait_asynccnt 0x0" ::: "memory");
        }
        __syncthreads();

        const _Float16* Ab = As[cur];
        const _Float16* Bb = Bst[cur];
#pragma unroll
        for (int kk = 0; kk < BK; kk += 32) {
            v16h a0, a1, b0, b1;
            const _Float16* ap0 = Ab + (wm * 32 + lr) * BK + kk + hi * 8;
            ((h8*)&a0)[0] = *(const h8*)(ap0);
            ((h8*)&a0)[1] = *(const h8*)(ap0 + 16);
            const _Float16* ap1 = Ab + (wm * 32 + 16 + lr) * BK + kk + hi * 8;
            ((h8*)&a1)[0] = *(const h8*)(ap1);
            ((h8*)&a1)[1] = *(const h8*)(ap1 + 16);
            const _Float16* bp0 = Bb + (wn * 32 + lr) * BK + kk + hi * 8;
            ((h8*)&b0)[0] = *(const h8*)(bp0);
            ((h8*)&b0)[1] = *(const h8*)(bp0 + 16);
            const _Float16* bp1 = Bb + (wn * 32 + 16 + lr) * BK + kk + hi * 8;
            ((h8*)&b1)[0] = *(const h8*)(bp1);
            ((h8*)&b1)[1] = *(const h8*)(bp1 + 16);

            acc[0][0] = __builtin_amdgcn_wmma_f32_16x16x32_f16(false, a0, false, b0, (short)0, acc[0][0], false, false);
            acc[0][1] = __builtin_amdgcn_wmma_f32_16x16x32_f16(false, a0, false, b1, (short)0, acc[0][1], false, false);
            acc[1][0] = __builtin_amdgcn_wmma_f32_16x16x32_f16(false, a1, false, b0, (short)0, acc[1][0], false, false);
            acc[1][1] = __builtin_amdgcn_wmma_f32_16x16x32_f16(false, a1, false, b1, (short)0, acc[1][1], false, false);
        }
        __syncthreads();   // protects buffer 'cur' from next iteration's stage
    }

    // --- epilogue: bias + ReLU + BN(eval), f16 store -----------------------
#pragma unroll
    for (int fm = 0; fm < 2; ++fm) {
#pragma unroll
        for (int fn = 0; fn < 2; ++fn) {
            int col = bn + wn * 32 + fn * 16 + lr;
            float bsv = bias[col];
            float ga  = gamma[col] * BN_INV;
            float bev = beta[col];
            v8f cc = acc[fm][fn];
#pragma unroll
            for (int v = 0; v < 8; ++v) {
                int row = bm + wm * 32 + fm * 16 + hi * 8 + v;
                if (row < M) {
                    float z = cc[v] + bsv;
                    z = fmaxf(z, 0.0f);
                    out[(size_t)row * Nc + col] = (_Float16)(ga * z + bev);
                }
            }
        }
    }
}

// ---------------------------------------------------------------------------
// masked max-pool over K neighbor slots -> f32
__global__ void k_maxpool(const _Float16* __restrict__ msg, const int* __restrict__ nbr,
                          int M, int C, float* __restrict__ out)
{
    size_t g = (size_t)blockIdx.x * blockDim.x + threadIdx.x;
    if (g >= (size_t)M * C) return;
    int m = (int)(g / C);
    int c = (int)(g - (size_t)m * C);
    float best = -3.4e38f;
    const int* row = nbr + (size_t)m * KNBR;
    for (int k = 0; k < KNBR; ++k) {
        if (row[k] >= 0) {
            float v = (float)msg[((size_t)m * KNBR + k) * C + c];
            best = fmaxf(best, v);
        }
    }
    out[g] = best;
}

// global max-pool over M rows -> f32 (1 x C)
__global__ void k_gpool(const _Float16* __restrict__ msg, int M, int C,
                        float* __restrict__ out)
{
    int c = blockIdx.x * blockDim.x + threadIdx.x;
    if (c >= C) return;
    float best = -3.4e38f;
    for (int m = 0; m < M; ++m) best = fmaxf(best, (float)msg[(size_t)m * C + c]);
    out[c] = best;
}

// ---------------------------------------------------------------------------
extern "C" void kernel_launch(void* const* d_in, const int* in_sizes, int n_in,
                              void* d_out, int out_size, void* d_ws, size_t ws_size,
                              hipStream_t stream)
{
    (void)in_sizes; (void)n_in; (void)out_size; (void)ws_size;
    const float* pos = (const float*)d_in[0];
    // params: (W,b,g,be) x3 per SA module
    const float *W_[9], *B_[9], *G_[9], *E_[9];
    for (int m = 0; m < 3; ++m)
        for (int l = 0; l < 3; ++l) {
            int base = 1 + m * 12 + l * 4;
            W_[m * 3 + l] = (const float*)d_in[base + 0];
            B_[m * 3 + l] = (const float*)d_in[base + 1];
            G_[m * 3 + l] = (const float*)d_in[base + 2];
            E_[m * 3 + l] = (const float*)d_in[base + 3];
        }

    float* out = (float*)d_out;
    const size_t OFF_X1 = 196608, OFF_C1 = 1874432, OFF_X2 = 1884263,
                 OFF_C2 = 2052199, OFF_X3 = 2052691;
    float* x1 = out + OFF_X1;  float* c1 = out + OFF_C1;
    float* x2 = out + OFF_X2;  float* c2 = out + OFF_C2;
    float* x3 = out + OFF_X3;

    // ---- workspace carve (256B aligned) -----------------------------------
    uint8_t* w = (uint8_t*)d_ws;
    size_t off = 0;
    auto carve = [&](size_t bytes) -> void* {
        void* p = w + off;
        off = (off + bytes + 255) & ~(size_t)255;
        return p;
    };
    float* dist1 = (float*)carve(NPTS * 4);
    int* idx1 = (int*)carve(M1 * 4);
    int* idx2 = (int*)carve(M2 * 4);
    int* nbr1 = (int*)carve((size_t)M1 * KNBR * 4);
    int* nbr2 = (int*)carve((size_t)M2 * KNBR * 4);
    // converted weights (f16, transposed [Cout][KaPad]), KaPad % 64 == 0
    const int KaPad[9] = {64, 128, 256, 576, 512, 1024, 1088, 1024, 2048};
    const int Cin_[9]  = {6, 128, 256, 515, 512, 1024, 1027, 1024, 2048};
    const int Cout[9]  = {128, 256, 512, 512, 1024, 1024, 1024, 2048, 2048};
    _Float16* Wh[9];
    for (int i = 0; i < 9; ++i) Wh[i] = (_Float16*)carve((size_t)KaPad[i] * Cout[i] * 2);
    const size_t E1 = (size_t)M1 * KNBR;                 // 209728 edge rows
    _Float16* feat1 = (_Float16*)carve(E1 * 64 * 2);
    _Float16* bufB1 = (_Float16*)carve(E1 * 256 * 2);
    _Float16* bufA1 = (_Float16*)carve(E1 * 512 * 2);    // arena, reused for SA2/SA3
    // SA2/SA3 scratch carved inside bufA1 (live only after SA1 pooling)
    uint8_t* arena = (uint8_t*)bufA1;
    _Float16* feat2 = (_Float16*)(arena);                // 10496 x 576 f16 ~ 12 MB
    _Float16* bufA2 = (_Float16*)(arena + (16u << 20));
    _Float16* bufB2 = (_Float16*)(arena + (40u << 20));
    _Float16* feat3 = (_Float16*)(arena + (64u << 20));
    _Float16* bufA3 = (_Float16*)(arena + (66u << 20));
    _Float16* bufB3 = (_Float16*)(arena + (68u << 20));

    const int TB = 256;
    auto cdiv = [](size_t a, size_t b) { return (unsigned)((a + b - 1) / b); };

    // ---- 0. x / pos outputs ----------------------------------------------
    k_copy_pos<<<cdiv(NPTS * 3, TB), TB, 0, stream>>>(pos, out);

    // ---- weight conversion + transpose (all up front) --------------------
    for (int i = 0; i < 9; ++i)
        k_convw_t<<<cdiv((size_t)KaPad[i] * Cout[i], TB), TB, 0, stream>>>(
            W_[i], Cin_[i], Cout[i], KaPad[i], Wh[i]);

    // ---- SA1 --------------------------------------------------------------
    k_fps<<<1, 1024, 0, stream>>>(pos, NPTS, M1, idx1, dist1);
    k_gather3<<<cdiv(M1 * 3, TB), TB, 0, stream>>>(pos, idx1, M1, c1);
    k_radius<<<cdiv(M1, TB), TB, 0, stream>>>(pos, NPTS, c1, idx1, M1, 0.04f, nbr1);
    k_feat<<<cdiv(E1 * 64, TB), TB, 0, stream>>>(pos, 3, pos, c1, nbr1, M1, 64, feat1);

    k_gemm_wmma<<<dim3(cdiv(E1, BM), 1), 512, 0, stream>>>(feat1, Wh[0], B_[0], G_[0], E_[0], bufA1, (int)E1, 64, 128);
    k_gemm_wmma<<<dim3(cdiv(E1, BM), 2), 512, 0, stream>>>(bufA1, Wh[1], B_[1], G_[1], E_[1], bufB1, (int)E1, 128, 256);
    k_gemm_wmma<<<dim3(cdiv(E1, BM), 4), 512, 0, stream>>>(bufB1, Wh[2], B_[2], G_[2], E_[2], bufA1, (int)E1, 256, 512);
    k_maxpool<<<cdiv((size_t)M1 * 512, TB), TB, 0, stream>>>(bufA1, nbr1, M1, 512, x1);

    // ---- SA2 --------------------------------------------------------------
    const size_t E2 = (size_t)M2 * KNBR;                 // 10496 edge rows
    k_fps<<<1, 1024, 0, stream>>>(c1, M1, M2, idx2, dist1);
    k_gather3<<<cdiv(M2 * 3, TB), TB, 0, stream>>>(c1, idx2, M2, c2);
    k_radius<<<cdiv(M2, TB), TB, 0, stream>>>(c1, M1, c2, idx2, M2, 0.16f, nbr2);
    k_feat<<<cdiv(E2 * 576, TB), TB, 0, stream>>>(x1, 512, c1, c2, nbr2, M2, 576, feat2);

    k_gemm_wmma<<<dim3(cdiv(E2, BM), 4), 512, 0, stream>>>(feat2, Wh[3], B_[3], G_[3], E_[3], bufA2, (int)E2, 576, 512);
    k_gemm_wmma<<<dim3(cdiv(E2, BM), 8), 512, 0, stream>>>(bufA2, Wh[4], B_[4], G_[4], E_[4], bufB2, (int)E2, 512, 1024);
    k_gemm_wmma<<<dim3(cdiv(E2, BM), 8), 512, 0, stream>>>(bufB2, Wh[5], B_[5], G_[5], E_[5], bufA2, (int)E2, 1024, 1024);
    k_maxpool<<<cdiv((size_t)M2 * 1024, TB), TB, 0, stream>>>(bufA2, nbr2, M2, 1024, x2);

    // ---- SA3 (global) -----------------------------------------------------
    k_feat3<<<cdiv((size_t)M2 * 1088, TB), TB, 0, stream>>>(x2, c2, feat3);
    k_gemm_wmma<<<dim3(cdiv(M2, BM), 8), 512, 0, stream>>>(feat3, Wh[6], B_[6], G_[6], E_[6], bufA3, M2, 1088, 1024);
    k_gemm_wmma<<<dim3(cdiv(M2, BM), 16), 512, 0, stream>>>(bufA3, Wh[7], B_[7], G_[7], E_[7], bufB3, M2, 1024, 2048);
    k_gemm_wmma<<<dim3(cdiv(M2, BM), 16), 512, 0, stream>>>(bufB3, Wh[8], B_[8], G_[8], E_[8], bufA3, M2, 2048, 2048);
    k_gpool<<<cdiv(2048, TB), TB, 0, stream>>>(bufA3, M2, 2048, x3);
}
